// DCNN_Advanced_82772609728876
// MI455X (gfx1250) — compile-verified, matrix operand order checked by reading, not run on hardware
//
#include <hip/hip_runtime.h>
#include <hip/hip_bf16.h>
#include <math.h>

// ---------------------------------------------------------------------------
// Problem constants (from the reference)
// ---------------------------------------------------------------------------
#define B_    64
#define S_    512
#define E_    64
#define CW1_  7
#define F1_   32
#define CW2_  5
#define F2_   64
#define KTOP_ 8
#define S0P   (S_ + 2 * (CW1_ - 1))    // 524  (padded input to conv1)
#define S1_   (S0P - CW1_ + 1)         // 518  (conv1 output length)
#define K1SEL 259                      // max(8, ceil(518/2))
#define S2P   (K1SEL + 2 * (CW2_ - 1)) // 267  (padded input to conv2)
#define S2_   (S2P - CW2_ + 1)         // 263  (conv2 output length)
#define E2_   (E_ / 2)                 // 32   (folded embedding columns)

typedef __attribute__((ext_vector_type(16))) __bf16 v16bf;
typedef __attribute__((ext_vector_type(8)))  __bf16 v8bf;
typedef __attribute__((ext_vector_type(8)))  float  v8f;

// ---------------------------------------------------------------------------
// K1: embedding gather + seq padding.  x0: (B, 524, E) f32, zero pad of 6.
// ---------------------------------------------------------------------------
__global__ void k_embed_pad(const int* __restrict__ tokens,
                            const float* __restrict__ emb,
                            float* __restrict__ x0) {
  int tid = blockIdx.x * blockDim.x + threadIdx.x;
  if (tid >= B_ * S0P * E_) return;
  int e  = tid % E_;
  int t  = tid / E_;
  int sp = t % S0P;
  int b  = t / S0P;
  float v = 0.0f;
  int s = sp - (CW1_ - 1);
  if (s >= 0 && s < S_) {
    int tok = tokens[b * S_ + s];
    v = emb[(long)tok * E_ + e];
  }
  x0[tid] = v;
}

// ---------------------------------------------------------------------------
// K2 (fused): conv1 (Cin=1, K=7) + tanh + full descending bitonic sort of the
// 518 outputs, keeping top-259 into the padded conv2 input x2p (pad = 4).
// Block = (b, e, group of 8 filters): the 524-float embedding row is loaded
// into LDS once and shared by all 8 waves; each wave owns one filter f.
// LDS: 2096 B row + 32 KB sort buffers.
// ---------------------------------------------------------------------------
__global__ void k_conv1_sort(const float* __restrict__ x0,
                             const float* __restrict__ W1,
                             const float* __restrict__ b1,
                             __bf16* __restrict__ x2p) {
  __shared__ float xrow[S0P];
  __shared__ float sm[8][1024];
  const int lane = threadIdx.x & 31;
  const int w    = threadIdx.x >> 5;   // wave id -> filter within group
  int idx = blockIdx.x;                // B_*E_*4 = 16384 blocks
  const int fg = idx & 3;  idx >>= 2;
  const int e  = idx & 63; idx >>= 6;
  const int b  = idx;
  const int f  = fg * 8 + w;

  // Stage the (b, :, e) embedding column once per block (x0 is L2-resident).
  for (int i = threadIdx.x; i < S0P; i += 256)
    xrow[i] = x0[((long)b * S0P + i) * E_ + e];
  __syncthreads();

  float wc[CW1_];
#pragma unroll
  for (int t = 0; t < CW1_; ++t) wc[t] = W1[(e * CW1_ + t) * F1_ + f];
  const float bias = b1[e * F1_ + f];

  // conv1 + tanh directly into the sort buffer, -inf pad to 1024.
  for (int i = lane; i < 1024; i += 32) {
    float v = -3.4e38f;
    if (i < S1_) {
      float acc = bias;
#pragma unroll
      for (int t = 0; t < CW1_; ++t) acc += xrow[i + t] * wc[t];
      v = tanhf(acc);
    }
    sm[w][i] = v;
  }
  __syncthreads();

  // Bitonic sort, overall DESCENDING (matches top_k value ordering).
  for (int k = 2; k <= 1024; k <<= 1) {
    for (int j = k >> 1; j > 0; j >>= 1) {
      for (int i = lane; i < 1024; i += 32) {
        int ixj = i ^ j;
        if (ixj > i) {
          float a = sm[w][i], c = sm[w][ixj];
          bool desc = ((i & k) == 0);
          if (desc ? (a < c) : (a > c)) { sm[w][i] = c; sm[w][ixj] = a; }
        }
      }
      __syncthreads();
    }
  }

  // Write top-259 (descending) into padded tensor at seq offset 4.
  const long outBase = (((long)b * S2P + (CW2_ - 1)) * E_ + e) * F1_ + f;
  for (int i = lane; i < K1SEL; i += 32)
    x2p[outBase + (long)i * (E_ * F1_)] = (__bf16)sm[w][i];
}

// ---------------------------------------------------------------------------
// K3: stage conv2 weights into per-lane WMMA B-fragment layout (bf16).
// Fragment (e, kk, nt) is 512 halfs: lane = elem/16, h = elem%16.
// K mapping (mirrors the A-matrix 16-bit layout): lanes 0-15 hold K {0..7,
// 16..23}, lanes 16-31 hold K {8..15, 24..31} within the 32-wide k-step.
// ---------------------------------------------------------------------------
__global__ void k_stage_bfrag(const float* __restrict__ W2,
                              __bf16* __restrict__ bfrag) {
  int tid = blockIdx.x * blockDim.x + threadIdx.x;
  if (tid >= E_ * CW2_ * 4 * 512) return;
  int elem = tid & 511;
  int t    = tid >> 9;
  int nt   = t & 3;  t >>= 2;
  int kk   = t % CW2_;
  int e    = t / CW2_;
  int lane = elem >> 4;
  int h    = elem & 15;
  int grp    = h >> 3;                       // 0: low half, 1: +16
  int within = h & 7;
  int klocal = ((lane >> 4) ? 8 : 0) + within + grp * 16;  // 0..31
  int col    = nt * 16 + (lane & 15);
  bfrag[tid] = (__bf16)W2[(((long)e * CW2_ + kk) * F1_ + klocal) * F2_ + col];
}

// ---------------------------------------------------------------------------
// K4: conv2 bf16 WMMA GEMM + bias + tanh + embedding-pair FOLD, fused.
// Block = 4 waves, one e-PAIR (e = 2*e2, 2*e2+1) x two M-tile pairs:
//   wave 0: e even, pair0   wave 1: e even, pair1
//   wave 2: e odd,  pair0   wave 3: e odd,  pair1
// Each wave owns two 16x64 output tiles (B fragments feed 2 WMMAs each; A
// rows for the next tap prefetched with global_prefetch_b8 at near scope).
// After tanh, odd-e waves push their tiles through 16 KB LDS and even-e
// waves write the averaged (folded) result -> x3h (B, 263, 32, 64) bf16.
// This halves the conv2 output traffic vs writing the unfolded tensor.
// ---------------------------------------------------------------------------
__global__ void k_conv2_wmma_fold(const __bf16* __restrict__ x2p,
                                  const __bf16* __restrict__ bfrag,
                                  const float* __restrict__ b2,
                                  __bf16* __restrict__ x3h) {
  __shared__ float xch[2][2][8][4][32];      // [pairSlot][p][j][nt][lane]
  const int lane  = threadIdx.x & 31;
  const int wid   = threadIdx.x >> 5;        // 0..3
  const int eLoc  = wid >> 1;                // 0: even e, 1: odd e
  const int pSlot = wid & 1;                 // pair slot within block
  const int pair  = blockIdx.x * 2 + pSlot;  // 0..525
  const int e2    = blockIdx.y;              // 0..31
  const int e     = e2 * 2 + eLoc;
  const int r     = lane & 15;
  const int hiSel = lane >> 4;               // lanes 16-31 take the +8 K half

  long rowBase[2];
#pragma unroll
  for (int p = 0; p < 2; ++p) {
    int m = (pair * 2 + p) * 16 + r;
    unsigned bb = (unsigned)m / 263u;
    unsigned ss = (unsigned)m - 263u * bb;
    rowBase[p] = (((long)bb * S2P + ss) * E_ + e) * F1_ + hiSel * 8;
  }

  v8f acc[2][4];
#pragma unroll
  for (int p = 0; p < 2; ++p)
#pragma unroll
    for (int nt = 0; nt < 4; ++nt)
#pragma unroll
      for (int i = 0; i < 8; ++i) acc[p][nt][i] = 0.0f;

  const __bf16* pbbase = bfrag + (long)e * (CW2_ * 4 * 512) + lane * 16;

#pragma unroll
  for (int kk = 0; kk < CW2_; ++kk) {
    v16bf a[2];
#pragma unroll
    for (int p = 0; p < 2; ++p) {
      const __bf16* pa = x2p + rowBase[p] + (long)kk * (E_ * F1_);
      v8bf alo = *(const v8bf*)(pa);
      v8bf ahi = *(const v8bf*)(pa + 16);
#pragma unroll
      for (int i = 0; i < 8; ++i) { a[p][i] = alo[i]; a[p][8 + i] = ahi[i]; }
      if (kk + 1 < CW2_)  // pull next tap's rows into the near cache
        __builtin_prefetch((const void*)(pa + (E_ * F1_)), 0, 3);
    }
#pragma unroll
    for (int nt = 0; nt < 4; ++nt) {
      v16bf bf = *(const v16bf*)(pbbase + (kk * 4 + nt) * 512);
      acc[0][nt] = __builtin_amdgcn_wmma_f32_16x16x32_bf16(
          false, a[0], false, bf, (short)0, acc[0][nt], false, false);
      acc[1][nt] = __builtin_amdgcn_wmma_f32_16x16x32_bf16(
          false, a[1], false, bf, (short)0, acc[1][nt], false, false);
    }
  }

  // Bias + tanh in place (C/D layout: row = hiSel*8 + j, col = lane&15).
  const int col = lane & 15;
  float bias[4];
#pragma unroll
  for (int nt = 0; nt < 4; ++nt) bias[nt] = b2[e * F2_ + nt * 16 + col];
#pragma unroll
  for (int p = 0; p < 2; ++p)
#pragma unroll
    for (int nt = 0; nt < 4; ++nt)
#pragma unroll
      for (int j = 0; j < 8; ++j)
        acc[p][nt][j] = tanhf(acc[p][nt][j] + bias[nt]);

  // Exchange odd-e tiles through LDS, then even-e waves write folded output.
  if (eLoc == 1) {
#pragma unroll
    for (int p = 0; p < 2; ++p)
#pragma unroll
      for (int j = 0; j < 8; ++j)
#pragma unroll
        for (int nt = 0; nt < 4; ++nt)
          xch[pSlot][p][j][nt][lane] = acc[p][nt][j];
  }
  __syncthreads();
  if (eLoc == 0) {
#pragma unroll
    for (int p = 0; p < 2; ++p) {
#pragma unroll
      for (int j = 0; j < 8; ++j) {
        int mo = (pair * 2 + p) * 16 + hiSel * 8 + j;
        unsigned bo = (unsigned)mo / 263u;
        unsigned so = (unsigned)mo - 263u * bo;
        long outBase = (((long)bo * S2_ + so) * E2_ + e2) * F2_;
#pragma unroll
        for (int nt = 0; nt < 4; ++nt) {
          float folded = 0.5f * (acc[p][nt][j] + xch[pSlot][p][j][nt][lane]);
          x3h[outBase + nt * 16 + col] = (__bf16)folded;
        }
      }
    }
  }
}

// ---------------------------------------------------------------------------
// K5: branch-free top-8 per (b, e2, f) row over the folded conv2 output.
// x5: (B, 8, 32, 64) f32 flattened as (B, 16384) in reference reshape order.
// ---------------------------------------------------------------------------
__global__ void k_top8(const __bf16* __restrict__ x3h,
                       float* __restrict__ x5) {
  int tid = blockIdx.x * blockDim.x + threadIdx.x;
  if (tid >= B_ * E2_ * F2_) return;
  int f  = tid % F2_;
  int t  = tid / F2_;
  int e2 = t % E2_;
  int b  = t / E2_;
  long base = (((long)b * S2_) * E2_ + e2) * F2_ + f;
  float top[KTOP_];
#pragma unroll
  for (int p = 0; p < KTOP_; ++p) top[p] = -3.4e38f;
  for (int s = 0; s < S2_; ++s) {
    float v = (float)x3h[base + (long)s * (E2_ * F2_)];
    // sorted-insert through 8 registers (descending), discard smallest
#pragma unroll
    for (int p = 0; p < KTOP_; ++p) {
      float hi = fmaxf(top[p], v);
      float lo = fminf(top[p], v);
      top[p] = hi;
      v = lo;
    }
  }
#pragma unroll
  for (int k = 0; k < KTOP_; ++k)
    x5[(long)b * (KTOP_ * E2_ * F2_) + (k * E2_ + e2) * F2_ + f] = top[k];
}

// ---------------------------------------------------------------------------
// K6: dense head: out[b] = sigmoid(dot(x5[b], Wd) + bd).  Block per batch row.
// ---------------------------------------------------------------------------
__global__ void k_dense_sigmoid(const float* __restrict__ x5,
                                const float* __restrict__ Wd,
                                const float* __restrict__ bd,
                                float* __restrict__ out) {
  __shared__ float red[256];
  const int b = blockIdx.x;
  const int tid = threadIdx.x;
  const int N = KTOP_ * E2_ * F2_;  // 16384
  float acc = 0.0f;
  for (int i = tid; i < N; i += 256)
    acc += x5[(long)b * N + i] * Wd[i];
  red[tid] = acc;
  __syncthreads();
  for (int st = 128; st > 0; st >>= 1) {
    if (tid < st) red[tid] += red[tid + st];
    __syncthreads();
  }
  if (tid == 0) out[b] = 1.0f / (1.0f + expf(-(red[0] + bd[0])));
}

// ---------------------------------------------------------------------------
// Launch
// ---------------------------------------------------------------------------
extern "C" void kernel_launch(void* const* d_in, const int* in_sizes, int n_in,
                              void* d_out, int out_size, void* d_ws, size_t ws_size,
                              hipStream_t stream) {
  const int*   tokens = (const int*)d_in[0];
  const float* emb    = (const float*)d_in[1];
  const float* W1     = (const float*)d_in[2];
  const float* b1     = (const float*)d_in[3];
  const float* W2     = (const float*)d_in[4];
  const float* b2     = (const float*)d_in[5];
  const float* Wd     = (const float*)d_in[6];
  const float* bd     = (const float*)d_in[7];
  float* out = (float*)d_out;

  // Workspace layout (all sizes naturally 256B aligned), ~153 MB total.
  char* ws = (char*)d_ws;
  size_t off = 0;
  float*  x0    = (float*)(ws + off);  off += (size_t)B_ * S0P * E_ * sizeof(float);           // 8.6 MB
  __bf16* x2p   = (__bf16*)(ws + off); off += (size_t)B_ * S2P * E_ * F1_ * sizeof(__bf16);    // 70.0 MB
  __bf16* bfrag = (__bf16*)(ws + off); off += (size_t)E_ * CW2_ * 4 * 512 * sizeof(__bf16);    // 1.3 MB
  __bf16* x3h   = (__bf16*)(ws + off); off += (size_t)B_ * S2_ * E2_ * F2_ * sizeof(__bf16);   // 69.0 MB
  float*  x5    = (float*)(ws + off);  off += (size_t)B_ * KTOP_ * E2_ * F2_ * sizeof(float);  // 4.2 MB
  (void)ws_size; (void)in_sizes; (void)n_in; (void)out_size;

  // Zero the conv2 padded activation tensor (pad regions stay 0).
  hipMemsetAsync(x2p, 0, (size_t)B_ * S2P * E_ * F1_ * sizeof(__bf16), stream);

  {  // K1: gather + pad
    int n = B_ * S0P * E_;
    k_embed_pad<<<(n + 255) / 256, 256, 0, stream>>>(tokens, emb, x0);
  }
  {  // K2: fused conv1 + tanh + sort-top259 -> padded x2p
    k_conv1_sort<<<B_ * E_ * 4, 256, 0, stream>>>(x0, W1, b1, x2p);
  }
  {  // K3: stage B fragments
    int n = E_ * CW2_ * 4 * 512;
    k_stage_bfrag<<<(n + 255) / 256, 256, 0, stream>>>(W2, bfrag);
  }
  {  // K4: conv2 GEMM (v_wmma_f32_16x16x32_bf16) + tanh + fold
    dim3 grid(263, E2_);                // 263 x 32 blocks, 4 waves each
    k_conv2_wmma_fold<<<grid, 128, 0, stream>>>(x2p, bfrag, b2, x3h);
  }
  {  // K5: top-8 over folded output
    int n = B_ * E2_ * F2_;             // 131072
    k_top8<<<(n + 255) / 256, 256, 0, stream>>>(x3h, x5);
  }
  {  // K6: dense + sigmoid
    k_dense_sigmoid<<<B_, 256, 0, stream>>>(x5, Wd, bd, out);
  }
}
